// FastSpacetimeAttention_43533788512837
// MI455X (gfx1250) — matrix operand order repaired and checked by reading, not compiled, not run on hardware
//
#include <hip/hip_runtime.h>
#include <hip/hip_bf16.h>

typedef _Float16 half_t;
typedef _Float16 v8h  __attribute__((ext_vector_type(8)));
typedef _Float16 v16h __attribute__((ext_vector_type(16)));
typedef float    v8f  __attribute__((ext_vector_type(8)));

#define B_SZ    8
#define DIM_    512
#define NTOK_   3137
#define ROWS_   (B_SZ*NTOK_)     /* 25096 */
#define G_CNT   1024             /* B*h*f */
#define NSP_    196
#define NKV_    197
#define MF_     266
#define MPAD_   288              /* 9 * 32 */
#define CT_     17               /* 272 / 16 column tiles for projection */

__device__ __forceinline__ v8f wmma16(v16h a, v16h b, v8f c) {
  return __builtin_amdgcn_wmma_f32_16x16x32_f16(false, a, false, b, (short)0, c, false, false);
}

__device__ __forceinline__ v8f vzero8() {
  v8f c;
#pragma unroll
  for (int i = 0; i < 8; ++i) c[i] = 0.f;
  return c;
}

// A fragment: 16x32 f16, row-major source. lane<16: row M=lane, K 0-15;
// lane>=16: row M=lane-16, K 16-31.
__device__ __forceinline__ v16h load_a_frag(const half_t* row, int k0, int lane) {
  int kb = (lane < 16) ? 0 : 16;
  v8h lo = *(const v8h*)(row + k0 + kb);
  v8h hi = *(const v8h*)(row + k0 + kb + 8);
  v16h a;
#pragma unroll
  for (int j = 0; j < 8; ++j) { a[j] = lo[j]; a[8 + j] = hi[j]; }
  return a;
}

// B fragment from a transposed (N-major) matrix Bt[n][k]: lane<16 col=lane K 0-15,
// lane>=16 col=lane-16 K 16-31 -> contiguous 16 halfs at colrow+k0+toff.
__device__ __forceinline__ v16h load_bt_frag(const half_t* colrow, int k0, int lane) {
  int toff = (lane < 16) ? 0 : 16;
  v8h lo = *(const v8h*)(colrow + k0 + toff);
  v8h hi = *(const v8h*)(colrow + k0 + toff + 8);
  v16h b;
#pragma unroll
  for (int j = 0; j < 8; ++j) { b[j] = lo[j]; b[8 + j] = hi[j]; }
  return b;
}

// ---------------- prep kernels ----------------
__global__ __launch_bounds__(256) void k_conv_f16(const float* __restrict__ src,
                                                  half_t* __restrict__ dst, int n) {
  int i = blockIdx.x * 256 + threadIdx.x;
  if (i < n) dst[i] = (half_t)src[i];
}

__global__ __launch_bounds__(256) void k_transpose_f16(const float* __restrict__ src,
                                                       half_t* __restrict__ dst, int R, int C) {
  int i = blockIdx.x * 256 + threadIdx.x;
  if (i < R * C) { int r = i / C, c = i - r * C; dst[(size_t)c * R + r] = (half_t)src[i]; }
}

// proj (266x64) -> f16 (272x64), scaled by dn = 64^-0.25 (folded into B operand), pad rows zero
__global__ __launch_bounds__(256) void k_prep_proj(const float* __restrict__ src,
                                                   half_t* __restrict__ dst) {
  int i = blockIdx.x * 256 + threadIdx.x;
  if (i < 272 * 64) {
    int r = i >> 6;
    dst[i] = (r < MF_) ? (half_t)(0.35355339059327373f * src[i]) : (half_t)0.f;
  }
}

// ---------------- stage 1: qkv GEMM ----------------
// A = x_f16 [25096 x 512], Bt = Wqkv^T f16 [1536 x 512]
// writes q/k/v in (b*h, n, 64) layout, q scaled by dh^-0.5
__global__ __launch_bounds__(256) void k_gemm_qkv(const half_t* __restrict__ A,
                                                  const half_t* __restrict__ Bt,
                                                  half_t* __restrict__ qb,
                                                  half_t* __restrict__ kb2,
                                                  half_t* __restrict__ vb) {
  int wave = threadIdx.x >> 5, lane = threadIdx.x & 31, rl = lane & 15;
  int tile = blockIdx.x * 8 + wave;            // 1569*96 tiles
  int mt = tile / 96, nt = tile - mt * 96;
  int row = mt * 16 + rl; if (row >= ROWS_) row = ROWS_ - 1;
  const half_t* arow = A + (size_t)row * DIM_;
  const half_t* bcol = Bt + (size_t)(nt * 16 + rl) * DIM_;
  v8f c = vzero8();
  for (int k0 = 0; k0 < DIM_; k0 += 32)
    c = wmma16(load_a_frag(arow, k0, lane), load_bt_frag(bcol, k0, lane), c);
  int col = nt * 16 + rl;
  int which = col >> 9, cc = col & 511, hh = cc >> 6, d = cc & 63;
  half_t* dstb = (which == 0) ? qb : ((which == 1) ? kb2 : vb);
  float sc = (which == 0) ? 0.125f : 1.0f;
  int rbase = mt * 16 + ((lane < 16) ? 0 : 8);
#pragma unroll
  for (int r = 0; r < 8; ++r) {
    int tok = rbase + r;
    if (tok < ROWS_) {
      int b = tok / NTOK_, n = tok - b * NTOK_;
      dstb[(((size_t)(b * 8 + hh)) * NTOK_ + n) * 64 + d] = (half_t)(c[r] * sc);
    }
  }
}

// ---------------- stage 2: cls-token softmax attention ----------------
__global__ __launch_bounds__(256) void k_cls(const half_t* __restrict__ qb,
                                             const half_t* __restrict__ kb,
                                             const half_t* __restrict__ vb,
                                             half_t* __restrict__ A2) {
  int bh = blockIdx.x, tid = threadIdx.x;
  __shared__ float clsq[64];
  __shared__ float red[256];
  __shared__ float accum[64];
  const half_t* qr = qb + (size_t)bh * NTOK_ * 64;
  const half_t* kr = kb + (size_t)bh * NTOK_ * 64;
  const half_t* vr = vb + (size_t)bh * NTOK_ * 64;
  if (tid < 64) { clsq[tid] = (float)qr[tid]; accum[tid] = 0.f; }
  __syncthreads();
  float mx = -3.0e38f;
  for (int j = tid; j < NTOK_; j += 256) {
    float s = 0.f;
    const half_t* krow = kr + (size_t)j * 64;
#pragma unroll
    for (int d = 0; d < 64; ++d) s += clsq[d] * (float)krow[d];
    mx = fmaxf(mx, s);
  }
  red[tid] = mx; __syncthreads();
  for (int s = 128; s > 0; s >>= 1) { if (tid < s) red[tid] = fmaxf(red[tid], red[tid + s]); __syncthreads(); }
  mx = red[0]; __syncthreads();
  float ssum = 0.f;
  float acc[64];
#pragma unroll
  for (int d = 0; d < 64; ++d) acc[d] = 0.f;
  for (int j = tid; j < NTOK_; j += 256) {
    float s = 0.f;
    const half_t* krow = kr + (size_t)j * 64;
#pragma unroll
    for (int d = 0; d < 64; ++d) s += clsq[d] * (float)krow[d];
    float e = __expf(s - mx);
    ssum += e;
    const half_t* vrow = vr + (size_t)j * 64;
#pragma unroll
    for (int d = 0; d < 64; ++d) acc[d] += e * (float)vrow[d];
  }
  red[tid] = ssum; __syncthreads();
  for (int s = 128; s > 0; s >>= 1) { if (tid < s) red[tid] += red[tid + s]; __syncthreads(); }
  float stot = red[0];
#pragma unroll
  for (int d = 0; d < 64; ++d) atomicAdd(&accum[d], acc[d]);
  __syncthreads();
  if (tid < 64) {
    int b = bh >> 3, hh = bh & 7;
    A2[((size_t)b * NTOK_) * 512 + hh * 64 + tid] = (half_t)(accum[tid] / stot);
  }
}

// ---------------- stage 3: performer feature projection ----------------
// MODE 0: queries -> qp (per-row stab, fused).  MODE 1: keys, global-max only.
// MODE 2: keys -> kp using global stab.
template <int MODE>
__global__ __launch_bounds__(256) void k_proj(const half_t* __restrict__ data_bh,
                                              const half_t* __restrict__ projh,
                                              half_t* __restrict__ outp,
                                              const float* __restrict__ stab_in,
                                              float* __restrict__ blockmax) {
  constexpr int ROWSG = (MODE == 0) ? NSP_ : NKV_;
  __shared__ float wred[8];
  int wave = threadIdx.x >> 5, lane = threadIdx.x & 31, rl = lane & 15;
  int wid = blockIdx.x * 8 + wave;          // G_CNT * 13 waves
  int g = wid / 13, mt = wid - g * 13;
  int bh = g >> 4, ff = g & 15;
  int n = mt * 16 + rl; if (n > ROWSG - 1) n = ROWSG - 1;
  int tok = (MODE == 0) ? (1 + ff * 196 + n) : ((n == 0) ? 0 : (1 + ff * 196 + n - 1));
  const half_t* arow = data_bh + ((size_t)bh * NTOK_ + tok) * 64;

  float diagv = 0.f;
  if constexpr (MODE != 1) {
    float sq = 0.f;
#pragma unroll
    for (int d = 0; d < 64; ++d) { float x = (float)arow[d]; sq += x * x; }
    diagv = 0.0625f * sq;                   // 0.5 * dn^2 * sum(x^2)
  }

  v8f cfr[CT_];
#pragma unroll
  for (int ct = 0; ct < CT_; ++ct) cfr[ct] = vzero8();
  for (int k0 = 0; k0 < 64; k0 += 32) {
    v16h a = load_a_frag(arow, k0, lane);
#pragma unroll
    for (int ct = 0; ct < CT_; ++ct) {
      int col = ct * 16 + rl;
      v16h b = load_bt_frag(projh + (size_t)col * 64, k0, lane);
      cfr[ct] = wmma16(a, b, cfr[ct]);
    }
  }

  float rmax[8];
#pragma unroll
  for (int r = 0; r < 8; ++r) rmax[r] = -3.0e38f;

  if constexpr (MODE == 0) {
#pragma unroll
    for (int ct = 0; ct < CT_; ++ct) {
      int col = ct * 16 + rl;
      if (col < MF_) {
#pragma unroll
        for (int r = 0; r < 8; ++r) rmax[r] = fmaxf(rmax[r], cfr[ct][r]);
      }
    }
#pragma unroll
    for (int off = 1; off < 16; off <<= 1) {
#pragma unroll
      for (int r = 0; r < 8; ++r) rmax[r] = fmaxf(rmax[r], __shfl_xor(rmax[r], off, 32));
    }
  }

  if constexpr (MODE == 1) {
    float wmax = -3.0e38f;
#pragma unroll
    for (int ct = 0; ct < CT_; ++ct) {
      int col = ct * 16 + rl;
      if (col < MF_) {
#pragma unroll
        for (int r = 0; r < 8; ++r) wmax = fmaxf(wmax, cfr[ct][r]);
      }
    }
#pragma unroll
    for (int off = 1; off < 32; off <<= 1) wmax = fmaxf(wmax, __shfl_xor(wmax, off, 32));
    if (lane == 0) wred[wave] = wmax;
    __syncthreads();
    if (threadIdx.x == 0) {
      float m = wred[0];
#pragma unroll
      for (int w = 1; w < 8; ++w) m = fmaxf(m, wred[w]);
      blockmax[blockIdx.x] = m;
    }
  } else {
    float dg[8];
#pragma unroll
    for (int r = 0; r < 8; ++r) dg[r] = __shfl(diagv, (lane < 16) ? r : (8 + r), 32);
    float stab_s = 0.f;
    if constexpr (MODE == 2) stab_s = stab_in[0];
    half_t* gout = outp + (size_t)g * ROWSG * MPAD_;
    int rowbase = mt * 16 + ((lane < 16) ? 0 : 8);
#pragma unroll
    for (int ct = 0; ct < CT_; ++ct) {
      int col = ct * 16 + rl;
#pragma unroll
      for (int r = 0; r < 8; ++r) {
        int rown = rowbase + r;
        if (rown < ROWSG) {
          float st = (MODE == 0) ? rmax[r] : stab_s;
          float val = 0.f;
          if (col < MF_) val = 0.061313933948f * (__expf(cfr[ct][r] - dg[r] - st) + 1e-4f);
          gout[(size_t)rown * MPAD_ + col] = (half_t)val;
        }
      }
    }
    if (lane < 16) {                         // zero K-pad cols 272..287
      int rown = mt * 16 + lane;
      if (rown < ROWSG) {
        v8h z;
#pragma unroll
        for (int j = 0; j < 8; ++j) z[j] = (half_t)0.f;
        *(v8h*)(gout + (size_t)rown * MPAD_ + 272) = z;
        *(v8h*)(gout + (size_t)rown * MPAD_ + 280) = z;
      }
    }
  }
}

__global__ __launch_bounds__(256) void k_reduce_max(const float* __restrict__ bm, int n,
                                                    float* __restrict__ stab) {
  __shared__ float red[256];
  float m = -3.0e38f;
  for (int i = threadIdx.x; i < n; i += 256) m = fmaxf(m, bm[i]);
  red[threadIdx.x] = m; __syncthreads();
  for (int s = 128; s > 0; s >>= 1) {
    if (threadIdx.x < s) red[threadIdx.x] = fmaxf(red[threadIdx.x], red[threadIdx.x + s]);
    __syncthreads();
  }
  if (threadIdx.x == 0) stab[0] = red[0];
}

// ---------------- stage 4: linear attention per group ----------------
__global__ __launch_bounds__(256) void k_groupattn(const half_t* __restrict__ qp,
                                                   const half_t* __restrict__ kp,
                                                   const half_t* __restrict__ vb,
                                                   half_t* __restrict__ A2) {
  int g = blockIdx.x, tid = threadIdx.x;
  int wave = tid >> 5, lane = tid & 31, rl = lane & 15;
  int bh = g >> 4, ff = g & 15, b = bh >> 3, hh = bh & 7;
  __shared__ half_t kp_t[32 * MPAD_];
  __shared__ half_t v_t[32 * 64];
  __shared__ half_t ctx[MPAD_ * 64];
  __shared__ float ksum[MPAD_];
  __shared__ float dinv[208];
  const half_t* kpg = kp + (size_t)g * NKV_ * MPAD_;
  const half_t* qpg = qp + (size_t)g * NSP_ * MPAD_;
  const half_t* vbh = vb + (size_t)bh * NTOK_ * 64;

  // Phase A: context[m][d] = sum_n kp[n][m] * v[n][d]   (288 x 64)
  v8f acc[9];
#pragma unroll
  for (int i = 0; i < 9; ++i) acc[i] = vzero8();
  for (int n0 = 0; n0 < 224; n0 += 32) {
    for (int idx = tid; idx < 32 * 144; idx += 256) {     // kp tile, dword units
      int r = idx / 144, c2 = (idx - r * 144) * 2;
      int nrow = n0 + r;
      unsigned int val = 0u;
      if (nrow < NKV_) val = *(const unsigned int*)(kpg + (size_t)nrow * MPAD_ + c2);
      *(unsigned int*)(&kp_t[r * MPAD_ + c2]) = val;
    }
    for (int idx = tid; idx < 32 * 32; idx += 256) {      // v tile, dword units
      int r = idx / 32, c2 = (idx - r * 32) * 2;
      int nrow = n0 + r;
      unsigned int val = 0u;
      if (nrow < NKV_) {
        int tokr = (nrow == 0) ? 0 : (1 + ff * 196 + nrow - 1);
        val = *(const unsigned int*)(vbh + (size_t)tokr * 64 + c2);
      }
      *(unsigned int*)(&v_t[r * 64 + c2]) = val;
    }
    __syncthreads();
#pragma unroll
    for (int i = 0; i < 9; ++i) {
      int t = wave * 9 + i;                 // 18 m-tiles x 4 d-tiles = 72
      int mtt = t >> 2, dt = t & 3;
      int m = mtt * 16 + rl;
      v16h a;
      int kb = (lane < 16) ? 0 : 16;
#pragma unroll
      for (int j = 0; j < 8; ++j) {
        a[j]     = kp_t[(kb + j) * MPAD_ + m];
        a[8 + j] = kp_t[(kb + 8 + j) * MPAD_ + m];
      }
      v16h bf;
      int toff = (lane < 16) ? 0 : 16;
      int d = dt * 16 + rl;
#pragma unroll
      for (int t2 = 0; t2 < 16; ++t2) bf[t2] = v_t[(toff + t2) * 64 + d];
      acc[i] = wmma16(a, bf, acc[i]);
    }
    __syncthreads();
  }
#pragma unroll
  for (int i = 0; i < 9; ++i) {
    int t = wave * 9 + i;
    int mtt = t >> 2, dt = t & 3;
    int d = dt * 16 + rl;
    int mbase = mtt * 16 + ((lane < 16) ? 0 : 8);
#pragma unroll
    for (int r = 0; r < 8; ++r) ctx[(mbase + r) * 64 + d] = (half_t)acc[i][r];
  }
  __syncthreads();

  // Phase B: k_sum and D_inv
  for (int m = tid; m < MPAD_; m += 256) {
    float s = 0.f;
    for (int nn = 0; nn < NKV_; ++nn) s += (float)kpg[(size_t)nn * MPAD_ + m];
    ksum[m] = s;
  }
  __syncthreads();
  if (tid < NSP_) {
    const half_t* qrow = qpg + (size_t)tid * MPAD_;
    float s = 0.f;
    for (int m = 0; m < MPAD_; ++m) s += (float)qrow[m] * ksum[m];
    dinv[tid] = 1.0f / s;
  }
  __syncthreads();

  // Phase C: out[n][d] = D_inv[n] * sum_m qp[n][m] * ctx[m][d]
#pragma unroll
  for (int i = 0; i < 7; ++i) {
    int t = wave + 8 * i;                   // 13 m-tiles x 4 d-tiles = 52
    if (t < 52) {
      int mtt = t >> 2, dt = t & 3;
      int rown = mtt * 16 + rl; if (rown > NSP_ - 1) rown = NSP_ - 1;
      const half_t* arow = qpg + (size_t)rown * MPAD_;
      v8f c2 = vzero8();
      for (int k0 = 0; k0 < MPAD_; k0 += 32) {
        v16h a = load_a_frag(arow, k0, lane);
        v16h bf;
        int toff = (lane < 16) ? 0 : 16;
        int d = dt * 16 + rl;
#pragma unroll
        for (int t2 = 0; t2 < 16; ++t2) bf[t2] = ctx[(k0 + toff + t2) * 64 + d];
        c2 = wmma16(a, bf, c2);
      }
      int d = dt * 16 + rl;
      int rbase = mtt * 16 + ((lane < 16) ? 0 : 8);
#pragma unroll
      for (int r = 0; r < 8; ++r) {
        int rn = rbase + r;
        if (rn < NSP_) {
          size_t tokg = (size_t)b * NTOK_ + 1 + ff * 196 + rn;
          A2[tokg * 512 + hh * 64 + d] = (half_t)(c2[r] * dinv[rn]);
        }
      }
    }
  }
}

// ---------------- stage 5: output projection ----------------
__global__ __launch_bounds__(256) void k_gemm_out(const half_t* __restrict__ A,
                                                  const half_t* __restrict__ Bt,
                                                  float* __restrict__ out) {
  int wave = threadIdx.x >> 5, lane = threadIdx.x & 31, rl = lane & 15;
  int tile = blockIdx.x * 8 + wave;         // 1569 * 32 tiles
  int mt = tile >> 5, nt = tile & 31;
  int row = mt * 16 + rl; if (row >= ROWS_) row = ROWS_ - 1;
  const half_t* arow = A + (size_t)row * DIM_;
  const half_t* bcol = Bt + (size_t)(nt * 16 + rl) * DIM_;
  v8f c = vzero8();
  for (int k0 = 0; k0 < DIM_; k0 += 32)
    c = wmma16(load_a_frag(arow, k0, lane), load_bt_frag(bcol, k0, lane), c);
  int col = nt * 16 + rl;
  int rbase = mt * 16 + ((lane < 16) ? 0 : 8);
#pragma unroll
  for (int r = 0; r < 8; ++r) {
    int tok = rbase + r;
    if (tok < ROWS_) out[(size_t)tok * DIM_ + col] = c[r];
  }
}

extern "C" void kernel_launch(void* const* d_in, const int* in_sizes, int n_in,
                              void* d_out, int out_size, void* d_ws, size_t ws_size,
                              hipStream_t stream) {
  const float* x    = (const float*)d_in[0];
  const float* Wqkv = (const float*)d_in[1];
  const float* Wout = (const float*)d_in[2];
  const float* proj = (const float*)d_in[3];
  float* out = (float*)d_out;
  char* ws = (char*)d_ws;
  size_t off = 0;
  auto alloc = [&](size_t bytes) { size_t o = off; off += (bytes + 255) & ~(size_t)255; return o; };

  half_t* xf   = (half_t*)(ws + alloc((size_t)ROWS_ * DIM_ * 2));
  half_t* wqt  = (half_t*)(ws + alloc((size_t)1536 * 512 * 2));
  half_t* wot  = (half_t*)(ws + alloc((size_t)512 * 512 * 2));
  half_t* prj  = (half_t*)(ws + alloc((size_t)272 * 64 * 2));
  half_t* qb   = (half_t*)(ws + alloc((size_t)64 * NTOK_ * 64 * 2));
  half_t* kb   = (half_t*)(ws + alloc((size_t)64 * NTOK_ * 64 * 2));
  half_t* vb   = (half_t*)(ws + alloc((size_t)64 * NTOK_ * 64 * 2));
  half_t* qp   = (half_t*)(ws + alloc((size_t)G_CNT * NSP_ * MPAD_ * 2));
  half_t* kp   = (half_t*)(ws + alloc((size_t)G_CNT * NKV_ * MPAD_ * 2));
  half_t* A2   = (half_t*)(ws + alloc((size_t)ROWS_ * DIM_ * 2));
  float*  bmax = (float*)(ws + alloc((size_t)1664 * 4));
  float*  stab = (float*)(ws + alloc(4));
  (void)in_sizes; (void)n_in; (void)out_size; (void)ws_size;

  k_conv_f16<<<(ROWS_ * DIM_ + 255) / 256, 256, 0, stream>>>(x, xf, ROWS_ * DIM_);
  k_transpose_f16<<<(512 * 1536 + 255) / 256, 256, 0, stream>>>(Wqkv, wqt, 512, 1536);
  k_transpose_f16<<<(512 * 512 + 255) / 256, 256, 0, stream>>>(Wout, wot, 512, 512);
  k_prep_proj<<<68, 256, 0, stream>>>(proj, prj);

  k_gemm_qkv<<<18828, 256, 0, stream>>>(xf, wqt, qb, kb, vb);   // 1569*96/8
  k_cls<<<64, 256, 0, stream>>>(qb, kb, vb, A2);

  k_proj<0><<<1664, 256, 0, stream>>>(qb, prj, qp, nullptr, nullptr);   // 1024*13/8
  k_proj<1><<<1664, 256, 0, stream>>>(kb, prj, nullptr, nullptr, bmax);
  k_reduce_max<<<1, 256, 0, stream>>>(bmax, 1664, stab);
  k_proj<2><<<1664, 256, 0, stream>>>(kb, prj, kp, stab, nullptr);

  k_groupattn<<<G_CNT, 256, 0, stream>>>(qp, kp, vb, A2);
  k_gemm_out<<<6276, 256, 0, stream>>>(A2, wot, out);           // 1569*32/8
}
